// GPTdecodeLayer_50929722196638
// MI455X (gfx1250) — matrix operand order repaired and checked by reading, not compiled, not run on hardware
//
#include <hip/hip_runtime.h>

#define B_    2
#define S_    2048
#define D_    1024
#define H_    16
#define DH_   64
#define SD_   (S_ * D_)          // 2,097,152
#define M_TOT (B_ * S_)          // 4096
#define QKV_LD 3072
#define EPS_  1e-5f

typedef __attribute__((ext_vector_type(16))) __bf16          v16bf;
typedef __attribute__((ext_vector_type(16))) unsigned short  v16us;
typedef __attribute__((ext_vector_type(8)))  float           v8f;
typedef __attribute__((ext_vector_type(8)))  unsigned int    v8u;
typedef int v4i_t __attribute__((vector_size(16)));

#if defined(__has_builtin)
# if __has_builtin(__builtin_amdgcn_global_load_async_to_lds_b128)
#  define HAVE_ASYNC_LDS 1
# endif
# if __has_builtin(__builtin_amdgcn_cvt_pk_bf16_f32)
#  define HAVE_CVT_PK 1
# endif
# if __has_builtin(__builtin_amdgcn_s_wait_asynccnt)
#  define HAVE_WAIT_ASYNC 1
# endif
#endif

__device__ __forceinline__ unsigned short f32_bf16(float f) {
  union { float f; unsigned u; } c; c.f = f;
  unsigned u = c.u;
  u += 0x7FFFu + ((u >> 16) & 1u);
  return (unsigned short)(u >> 16);
}
__device__ __forceinline__ unsigned pack_bf16(float lo, float hi) {
#ifdef HAVE_CVT_PK
  auto r = __builtin_amdgcn_cvt_pk_bf16_f32(lo, hi);   // v_cvt_pk_bf16_f32
  return __builtin_bit_cast(unsigned, r);
#else
  return (unsigned)f32_bf16(lo) | ((unsigned)f32_bf16(hi) << 16);
#endif
}
__device__ __forceinline__ v8f v8f_zero() {
  v8f z;
#pragma unroll
  for (int i = 0; i < 8; ++i) z[i] = 0.f;
  return z;
}

// ---- CDNA5 async global->LDS staging (ASYNCcnt path) ----------------------
__device__ __forceinline__ void async_b128(const void* g, void* l) {
#ifdef HAVE_ASYNC_LDS
  __builtin_amdgcn_global_load_async_to_lds_b128(
      (__attribute__((address_space(1))) v4i_t*)(g),
      (__attribute__((address_space(3))) v4i_t*)(l), 0, 0);
#else
  *(float4*)l = *(const float4*)g;
#endif
}
__device__ __forceinline__ void wait_async() {
#ifdef HAVE_ASYNC_LDS
# ifdef HAVE_WAIT_ASYNC
  __builtin_amdgcn_s_wait_asynccnt(0);
# else
  asm volatile("s_wait_asynccnt 0" ::: "memory");
# endif
#endif
}

// ---------------------------------------------------------------------------
// One-shot conversions
// ---------------------------------------------------------------------------
__global__ __launch_bounds__(256) void cvt_f32_bf16(
    const float* __restrict__ in, unsigned* __restrict__ out, int n4)
{
  int i = blockIdx.x * 256 + threadIdx.x;
  if (i < n4) {
    float4 f = ((const float4*)in)[i];
    out[2 * i]     = pack_bf16(f.x, f.y);
    out[2 * i + 1] = pack_bf16(f.z, f.w);
  }
}

// W fp32 [K][N] -> transposed pair-major u32 [N][K/2]
__global__ __launch_bounds__(256) void pack_weight_t(
    const float* __restrict__ W, unsigned* __restrict__ Wp, int K, int N)
{
  int idx = blockIdx.x * 256 + threadIdx.x;   // over N*(K/8)
  int n   = idx / (K / 8);
  int kq  = (idx % (K / 8)) * 4;
  if (n < N) {
    unsigned* o = Wp + (size_t)n * (K / 2) + kq;
#pragma unroll
    for (int j = 0; j < 4; ++j) {
      int kp = kq + j;
      o[j] = pack_bf16(W[(size_t)(2 * kp + 0) * N + n],
                       W[(size_t)(2 * kp + 1) * N + n]);
    }
  }
}

// V rows of bf16 qkv -> key-pair-packed u32: vpk[b][key/2][h*64+d]
__global__ __launch_bounds__(256) void repack_v(
    const unsigned short* __restrict__ qkv, unsigned* __restrict__ vpk)
{
  int idx = blockIdx.x * 256 + threadIdx.x;       // over B*(S/2)*D
  int d   = idx & (D_ - 1);
  int kp  = (idx >> 10) & (S_ / 2 - 1);
  int b   = idx >> 20;
  int h   = d >> 6, dd = d & 63;
  const unsigned short* src =
      qkv + (size_t)(b * S_ + 2 * kp) * QKV_LD + h * 192 + 128 + dd;
  unsigned lo = src[0];
  unsigned hi = src[QKV_LD];
  vpk[(size_t)(b * (S_ / 2) + kp) * D_ + d] = lo | (hi << 16);
}

// ---------------------------------------------------------------------------
// bf16 WMMA GEMM, async double-buffered LDS staging.
// ---------------------------------------------------------------------------
template <int HAS_RES, int RELU, int OUT_F32, int OUT_BF16>
__global__ __launch_bounds__(256) void gemm_wmma(
    const unsigned short* __restrict__ Abf,
    const unsigned* __restrict__ Wp,
    const float* __restrict__ bias, const float* __restrict__ res,
    float* __restrict__ Cf, unsigned short* __restrict__ Cb,
    int N, int K, int ldc)
{
  __shared__ __align__(16) unsigned a_lds[2][128][20];  // row stride 80B
  __shared__ __align__(16) unsigned b_lds[2][128][20];  // [col][kp]

  const int tid  = threadIdx.x;
  const int lane = tid & 31;
  const int wave = tid >> 5;
  const int half = lane >> 4;
  const int ln   = lane & 15;
  const int wm   = wave >> 2;
  const int wn   = wave & 3;
  const int gm0  = blockIdx.y * 128;
  const int gn0  = blockIdx.x * 128;

  auto stage = [&](int buf, int k0) {
#pragma unroll
    for (int t = 0; t < 2; ++t) {
      int idx = tid + 256 * t;
      int row = idx >> 2, q = idx & 3;
      async_b128(Abf + (size_t)(gm0 + row) * K + k0 + q * 8,
                 &a_lds[buf][row][q * 4]);
    }
#pragma unroll
    for (int t = 0; t < 2; ++t) {
      int idx = tid + 256 * t;
      int col = idx >> 2, q = idx & 3;
      async_b128(Wp + (size_t)(gn0 + col) * (K / 2) + k0 / 2 + q * 4,
                 &b_lds[buf][col][q * 4]);
    }
  };

  v8f acc[4][2];
#pragma unroll
  for (int i = 0; i < 4; ++i)
#pragma unroll
    for (int j = 0; j < 2; ++j) acc[i][j] = v8f_zero();

  const int nk = K >> 5;
  stage(0, 0);
  for (int t = 0; t < nk; ++t) {
    const int buf = t & 1;
    wait_async();
    __syncthreads();
    if (t + 1 < nk) stage(buf ^ 1, (t + 1) * 32);

    v16bf afr[4], bfr[2];
#pragma unroll
    for (int fm = 0; fm < 4; ++fm) {
      v8u u;
      int row = wm * 64 + fm * 16 + ln;
#pragma unroll
      for (int v = 0; v < 8; ++v) {
        int kp = (v < 4 ? v : v + 4) + 4 * half;
        u[v] = a_lds[buf][row][kp];
      }
      afr[fm] = __builtin_bit_cast(v16bf, u);
    }
#pragma unroll
    for (int fn = 0; fn < 2; ++fn) {
      v8u u;
      int col = wn * 32 + fn * 16 + ln;
#pragma unroll
      for (int v = 0; v < 8; ++v) u[v] = b_lds[buf][col][(v < 4 ? v : v + 4) + 4 * half];
      bfr[fn] = __builtin_bit_cast(v16bf, u);
    }
#pragma unroll
    for (int fm = 0; fm < 4; ++fm)
#pragma unroll
      for (int fn = 0; fn < 2; ++fn)
        acc[fm][fn] = __builtin_amdgcn_wmma_f32_16x16x32_bf16(
            false, afr[fm], false, bfr[fn], (short)0, acc[fm][fn], false, false);
    __syncthreads();
  }

#pragma unroll
  for (int fm = 0; fm < 4; ++fm) {
#pragma unroll
    for (int fn = 0; fn < 2; ++fn) {
      int col  = gn0 + wn * 32 + fn * 16 + ln;
      int row0 = gm0 + wm * 64 + fm * 16 + half * 8;
      float bv = bias[col];
#pragma unroll
      for (int r = 0; r < 8; ++r) {
        int row = row0 + r;
        float v = acc[fm][fn][r] + bv;
        if constexpr (HAS_RES) v += res[(size_t)row * ldc + col];
        if constexpr (RELU)    v = fmaxf(v, 0.f);
        if constexpr (OUT_F32)  Cf[(size_t)row * ldc + col] = v;
        if constexpr (OUT_BF16) Cb[(size_t)row * ldc + col] = f32_bf16(v);
      }
    }
  }
}

// ---------------------------------------------------------------------------
// Flash attention. exp2-domain softmax (1/sqrt(dh)*log2e folded into scale).
// Only the diagonal key tile applies the causal mask.
// ---------------------------------------------------------------------------
#define SCALE2_ 0.18033688011112042f   // 0.125 * log2(e)

template <bool MASKED>
__device__ __forceinline__ void attn_tile(
    int key0, int qrow0, int half, int ln,
    const unsigned short* __restrict__ kb, const unsigned* __restrict__ vp,
    unsigned short (*pw)[33],
    const v16bf qfr[2], float m_i[8], float l_i[8], v8f o[4])
{
  // ---- scores S[16x32] = Q @ K^T ----
  v8f s[2];
#pragma unroll
  for (int f = 0; f < 2; ++f) {
    s[f] = v8f_zero();
#pragma unroll
    for (int ks = 0; ks < 2; ++ks) {
      v8u u;
#pragma unroll
      for (int v = 0; v < 8; ++v) {
        int dd = 2 * v + (v >= 4 ? 8 : 0) + 8 * half + 32 * ks;
        u[v] = *(const unsigned*)(kb + (size_t)(key0 + f * 16 + ln) * QKV_LD + dd);
      }
      v16bf kfr = __builtin_bit_cast(v16bf, u);
      s[f] = __builtin_amdgcn_wmma_f32_16x16x32_bf16(
          false, qfr[ks], false, kfr, (short)0, s[f], false, false);
    }
  }

  // ---- online softmax (exp2 domain) ----
#pragma unroll
  for (int r = 0; r < 8; ++r) {
    float s0 = s[0][r] * SCALE2_, s1 = s[1][r] * SCALE2_;
    if constexpr (MASKED) {
      int qrow = qrow0 + r + 8 * half;
      if (key0 + ln      > qrow) s0 = -1e30f;
      if (key0 + 16 + ln > qrow) s1 = -1e30f;
    }
    float rmax = fmaxf(s0, s1);
#pragma unroll
    for (int off = 1; off < 16; off <<= 1)
      rmax = fmaxf(rmax, __shfl_xor(rmax, off, 32));
    float mold = m_i[r];
    float mnew = fmaxf(mold, rmax);
    float corr = exp2f(mold - mnew);
    float p0, p1;
    if constexpr (MASKED) {
      p0 = (s0 < -1e29f) ? 0.f : exp2f(s0 - mnew);
      p1 = (s1 < -1e29f) ? 0.f : exp2f(s1 - mnew);
    } else {
      p0 = exp2f(s0 - mnew);
      p1 = exp2f(s1 - mnew);
    }
    float rsum = p0 + p1;
#pragma unroll
    for (int off = 1; off < 16; off <<= 1) rsum += __shfl_xor(rsum, off, 32);
    l_i[r] = l_i[r] * corr + rsum;
    m_i[r] = mnew;
#pragma unroll
    for (int fn = 0; fn < 4; ++fn) o[fn][r] *= corr;
    pw[r + 8 * half][ln]      = f32_bf16(p0);
    pw[r + 8 * half][16 + ln] = f32_bf16(p1);
  }
  __builtin_amdgcn_wave_barrier();
  asm volatile("s_wait_dscnt 0" ::: "memory");

  // ---- regather P in A-fragment layout ----
  v16us pu;
#pragma unroll
  for (int v = 0; v < 8; ++v) {
    int kbase = 2 * v + (v >= 4 ? 8 : 0) + 8 * half;
    pu[2 * v]     = pw[ln][kbase];
    pu[2 * v + 1] = pw[ln][kbase + 1];
  }
  v16bf pfr = __builtin_bit_cast(v16bf, pu);
  __builtin_amdgcn_wave_barrier();
  asm volatile("s_wait_dscnt 0" ::: "memory");

  // ---- O += P @ V ----
#pragma unroll
  for (int fn = 0; fn < 4; ++fn) {
    v8u u;
#pragma unroll
    for (int v = 0; v < 8; ++v) {
      int kp = key0 / 2 + (v < 4 ? v : v + 4) + 4 * half;
      u[v] = vp[(size_t)kp * D_ + fn * 16 + ln];
    }
    v16bf vfr = __builtin_bit_cast(v16bf, u);
    o[fn] = __builtin_amdgcn_wmma_f32_16x16x32_bf16(
        false, pfr, false, vfr, (short)0, o[fn], false, false);
  }
}

__global__ __launch_bounds__(256) void attn_flash(
    const unsigned short* __restrict__ qkv,
    const unsigned* __restrict__ vpk,
    unsigned short* __restrict__ ctx)
{
  __shared__ unsigned short plds[8][16][33];

  const int tid  = threadIdx.x;
  const int lane = tid & 31;
  const int wave = tid >> 5;
  const int half = lane >> 4;
  const int ln   = lane & 15;
  const int b    = blockIdx.y >> 4;
  const int h    = blockIdx.y & 15;
  const int qrow0 = blockIdx.x * 128 + wave * 16;

  const unsigned short* qb = qkv + (size_t)b * S_ * QKV_LD + h * 192;
  const unsigned short* kb = qb + 64;
  const unsigned* vp = vpk + (size_t)b * (S_ / 2) * D_ + h * 64;

  v16bf qfr[2];
#pragma unroll
  for (int ks = 0; ks < 2; ++ks) {
    v8u u;
#pragma unroll
    for (int v = 0; v < 8; ++v) {
      int kbase = 2 * v + (v >= 4 ? 8 : 0) + 8 * half + 32 * ks;
      u[v] = *(const unsigned*)(qb + (size_t)(qrow0 + ln) * QKV_LD + kbase);
    }
    qfr[ks] = __builtin_bit_cast(v16bf, u);
  }

  float m_i[8], l_i[8];
  v8f o[4];
#pragma unroll
  for (int r = 0; r < 8; ++r) { m_i[r] = -1e30f; l_i[r] = 0.f; }
#pragma unroll
  for (int fn = 0; fn < 4; ++fn) o[fn] = v8f_zero();

  const int nfull = qrow0 >> 5;              // tiles fully below the diagonal
  for (int kt = 0; kt < nfull; ++kt)
    attn_tile<false>(kt * 32, qrow0, half, ln, kb, vp, plds[wave],
                     qfr, m_i, l_i, o);
  attn_tile<true>(nfull * 32, qrow0, half, ln, kb, vp, plds[wave],
                  qfr, m_i, l_i, o);        // single diagonal (masked) tile

  unsigned short* cb = ctx + (size_t)b * S_ * D_ + h * 64;
#pragma unroll
  for (int r = 0; r < 8; ++r) {
    int qrow = qrow0 + r + 8 * half;
    float inv = 1.f / fmaxf(l_i[r], 1e-20f);
#pragma unroll
    for (int fn = 0; fn < 4; ++fn)
      cb[(size_t)qrow * D_ + fn * 16 + ln] = f32_bf16(o[fn][r] * inv);
  }
}

// ---------------------------------------------------------------------------
// Global LayerNorm (mean/var over the whole S*D slab per batch)
// ---------------------------------------------------------------------------
__global__ void zero_stats(float* stats) {
  if (threadIdx.x < 4) stats[threadIdx.x] = 0.f;
}

__global__ __launch_bounds__(256) void ln_reduce(
    const float* __restrict__ x, float* __restrict__ stats)
{
  const int b = blockIdx.y;
  const float* xb = x + (size_t)b * SD_;
  int base = blockIdx.x * 2048 + threadIdx.x;
  float s = 0.f, q = 0.f;
#pragma unroll
  for (int i = 0; i < 8; ++i) {
    float v = xb[base + i * 256];
    s += v; q += v * v;
  }
#pragma unroll
  for (int off = 1; off < 32; off <<= 1) {
    s += __shfl_xor(s, off, 32);
    q += __shfl_xor(q, off, 32);
  }
  __shared__ float ls[8], lq[8];
  int wave = threadIdx.x >> 5;
  if ((threadIdx.x & 31) == 0) { ls[wave] = s; lq[wave] = q; }
  __syncthreads();
  if (threadIdx.x == 0) {
    float S = 0.f, Q = 0.f;
    for (int i = 0; i < 8; ++i) { S += ls[i]; Q += lq[i]; }
    atomicAdd(&stats[2 * b],     S);
    atomicAdd(&stats[2 * b + 1], Q);
  }
}

__global__ __launch_bounds__(256) void ln_apply(
    const float* __restrict__ x, const float* __restrict__ stats,
    const float* __restrict__ gamma, const float* __restrict__ beta,
    float* __restrict__ outf, unsigned short* __restrict__ outb)
{
  const int b = blockIdx.y;
  int e = blockIdx.x * 256 + threadIdx.x;
  float mean = stats[2 * b] * (1.f / (float)SD_);
  float var  = stats[2 * b + 1] * (1.f / (float)SD_) - mean * mean;
  float rstd = rsqrtf(var + EPS_);
  size_t idx = (size_t)b * SD_ + e;
  float v = (x[idx] - mean) * rstd * gamma[e] + beta[e];
  outf[idx] = v;
  if (outb) outb[idx] = f32_bf16(v);
}

// ---------------------------------------------------------------------------
extern "C" void kernel_launch(void* const* d_in, const int* in_sizes, int n_in,
                              void* d_out, int out_size, void* d_ws, size_t ws_size,
                              hipStream_t stream)
{
  const float* x   = (const float*)d_in[0];
  const float* Wq  = (const float*)d_in[1];
  const float* bq  = (const float*)d_in[2];
  const float* Wkv = (const float*)d_in[3];
  const float* bkv = (const float*)d_in[4];
  const float* Wo  = (const float*)d_in[5];
  const float* bo  = (const float*)d_in[6];
  const float* g1  = (const float*)d_in[7];
  const float* be1 = (const float*)d_in[8];
  const float* W1  = (const float*)d_in[9];
  const float* b1  = (const float*)d_in[10];
  const float* W2  = (const float*)d_in[11];
  const float* b2  = (const float*)d_in[12];
  const float* g2  = (const float*)d_in[13];
  const float* be2 = (const float*)d_in[14];
  float* out = (float*)d_out;

  char* ws = (char*)d_ws;
  size_t off = 0;
  auto take = [&](size_t bytes) { void* p = ws + off; off += (bytes + 255) & ~(size_t)255; return p; };

  unsigned short* xbf   = (unsigned short*)take((size_t)M_TOT * D_ * 2);
  unsigned*       wqp   = (unsigned*)take((size_t)D_ * (D_/2) * 4);
  unsigned*       wkvp  = (unsigned*)take((size_t)(2*D_) * (D_/2) * 4);
  unsigned*       wop   = (unsigned*)take((size_t)D_ * (D_/2) * 4);
  unsigned*       w1p   = (unsigned*)take((size_t)(2*D_) * (D_/2) * 4);
  unsigned*       w2p   = (unsigned*)take((size_t)D_ * D_ * 4);
  unsigned short* qkvbf = (unsigned short*)take((size_t)M_TOT * QKV_LD * 2);
  unsigned*       vpk   = (unsigned*)take((size_t)B_ * (S_/2) * D_ * 4);
  unsigned short* ctxbf = (unsigned short*)take((size_t)M_TOT * D_ * 2);
  float*          attn  = (float*)take((size_t)M_TOT * D_ * 4);
  float*          ybuf  = (float*)take((size_t)M_TOT * D_ * 4);
  unsigned short* ybf   = (unsigned short*)take((size_t)M_TOT * D_ * 2);
  unsigned short* hbf   = (unsigned short*)take((size_t)M_TOT * 2*D_ * 2);
  float*          stats = (float*)take(64);
  float*          yz    = attn;

  dim3 blk(256);

  cvt_f32_bf16<<<(M_TOT * D_ / 4 + 255) / 256, blk, 0, stream>>>(x, (unsigned*)xbf, M_TOT * D_ / 4);
  pack_weight_t<<<(D_   * (D_/8)   + 255) / 256, blk, 0, stream>>>(Wq,  wqp,  D_,   D_);
  pack_weight_t<<<(2*D_ * (D_/8)   + 255) / 256, blk, 0, stream>>>(Wkv, wkvp, D_,   2*D_);
  pack_weight_t<<<(D_   * (D_/8)   + 255) / 256, blk, 0, stream>>>(Wo,  wop,  D_,   D_);
  pack_weight_t<<<(2*D_ * (D_/8)   + 255) / 256, blk, 0, stream>>>(W1,  w1p,  D_,   2*D_);
  pack_weight_t<<<(D_   * (2*D_/8) + 255) / 256, blk, 0, stream>>>(W2,  w2p,  2*D_, D_);

  gemm_wmma<0,0,0,1><<<dim3(1024/128, M_TOT/128), blk, 0, stream>>>(
      xbf, wqp, bq, nullptr, nullptr, qkvbf, 1024, 1024, QKV_LD);
  gemm_wmma<0,0,0,1><<<dim3(2048/128, M_TOT/128), blk, 0, stream>>>(
      xbf, wkvp, bkv, nullptr, nullptr, qkvbf + 1024, 2048, 1024, QKV_LD);

  repack_v<<<(B_ * (S_/2) * D_) / 256, blk, 0, stream>>>(qkvbf, vpk);
  attn_flash<<<dim3(S_/128, B_*H_), blk, 0, stream>>>(qkvbf, vpk, ctxbf);

  gemm_wmma<1,0,1,0><<<dim3(1024/128, M_TOT/128), blk, 0, stream>>>(
      ctxbf, wop, bo, x, attn, nullptr, 1024, 1024, 1024);

  zero_stats<<<1, 64, 0, stream>>>(stats);
  ln_reduce<<<dim3(SD_/2048, B_), blk, 0, stream>>>(attn, stats);
  ln_apply<<<dim3(SD_/256, B_), blk, 0, stream>>>(attn, stats, g1, be1, ybuf, ybf);

  gemm_wmma<0,1,0,1><<<dim3(2048/128, M_TOT/128), blk, 0, stream>>>(
      ybf, w1p, b1, nullptr, nullptr, hbf, 2048, 1024, 2048);

  gemm_wmma<1,0,1,0><<<dim3(1024/128, M_TOT/128), blk, 0, stream>>>(
      hbf, w2p, b2, ybuf, yz, nullptr, 1024, 2048, 1024);

  zero_stats<<<1, 64, 0, stream>>>(stats);
  ln_reduce<<<dim3(SD_/2048, B_), blk, 0, stream>>>(yz, stats);
  ln_apply<<<dim3(SD_/256, B_), blk, 0, stream>>>(yz, stats, g2, be2, out, nullptr);
}
